// GINO_83270825935649
// MI455X (gfx1250) — compile-verified, matrix operand order checked by reading, not run on hardware
//
#include <hip/hip_runtime.h>
#include <hip/hip_bf16.h>
#include <math.h>
#include <stdint.h>

typedef __attribute__((ext_vector_type(16))) _Float16 v16h;
typedef __attribute__((ext_vector_type(8)))  float    v8f;
typedef int v4i_t __attribute__((vector_size(16)));

#define NQ 16384
#define QC 1024
#define PPC (QC * 27)          // 27648 pair-rows per GNO chunk
#define TWO_PI 6.283185307179586f

#if defined(__HIP_DEVICE_COMPILE__)
#if __has_builtin(__builtin_amdgcn_global_load_async_to_lds_b128) && \
    __has_builtin(__builtin_amdgcn_s_wait_asynccnt)
#define USE_ASYNC_LDS 1
#endif
#endif
#ifndef USE_ASYNC_LDS
#define USE_ASYNC_LDS 0
#endif

__device__ __forceinline__ float gelu_f(float v) {
    const float c0 = 0.7978845608028654f; // sqrt(2/pi)
    return 0.5f * v * (1.0f + tanhf(c0 * (v + 0.044715f * v * v * v)));
}

// 16-byte global -> LDS move; async (ASYNCcnt) on CDNA5 when available.
__device__ __forceinline__ void copy16_g2l(const _Float16* g, _Float16* l) {
#if USE_ASYNC_LDS
    __builtin_amdgcn_global_load_async_to_lds_b128(
        (__attribute__((address_space(1))) v4i_t*)(uintptr_t)g,
        (__attribute__((address_space(3))) v4i_t*)(uintptr_t)l, 0, 0);
#else
    *(uint4*)l = *(const uint4*)g;
#endif
}
__device__ __forceinline__ void g2l_wave_fence() {
#if USE_ASYNC_LDS
    __builtin_amdgcn_s_wait_asynccnt(0);
#endif
}

// ---------------------------------------------------------------------------
// Tiled WMMA GEMM:  C[M,N] = epilogue( A[M,K] @ W[N,K]^T )
// A, W are f16 row-major.  Requires M%128==0, N%64==0, K%32==0, 16B alignment.
// Block = 256 threads = 8 waves laid out 4(M) x 2(N); wave tile 32x32
// (2 A-frags x 2 B-frags -> 4 WMMAs per K-step); block tile 128x64.
// Double-buffered LDS; tile k+1 streams in via async-to-LDS while tile k
// is consumed by WMMA.
// ---------------------------------------------------------------------------
__global__ __launch_bounds__(256) void gemm_wmma(
    const _Float16* __restrict__ A, const _Float16* __restrict__ W,
    const float* __restrict__ bias, const float* __restrict__ Cin,
    float* __restrict__ Cout, _Float16* __restrict__ Cout16,
    int M, int N, int K, int do_gelu)
{
    __shared__ __align__(16) _Float16 As[2][128 * 32];
    __shared__ __align__(16) _Float16 Bs[2][64 * 32];

    const int tid  = threadIdx.x;
    const int lane = tid & 31;
    const int wid  = tid >> 5;     // 0..7
    const int wm   = wid >> 1;     // 0..3  (M sub-tile)
    const int wn   = wid & 1;      // 0..1  (N sub-tile)
    const int m0   = blockIdx.y * 128;
    const int n0   = blockIdx.x * 64;

    // cooperative-fill coordinates: each thread moves three 16B chunks
    const int ar = tid >> 2;             // 0..63
    const int ac = (tid & 3) * 8;        // half offset within 32-wide row

    const _Float16* gA0 = A + (size_t)(m0 + ar) * K + ac;
    const _Float16* gA1 = A + (size_t)(m0 + ar + 64) * K + ac;
    const _Float16* gW  = W + (size_t)(n0 + ar) * K + ac;
    const int lA0 = ar * 32 + ac;
    const int lA1 = (ar + 64) * 32 + ac;

    v8f acc00 = {}, acc01 = {}, acc10 = {}, acc11 = {};

    const int half_sel = (lane >> 4) << 3;           // 0 | 8
    const int lidx     = lane & 15;
    const int am0 = (wm * 32 + lidx) * 32;
    const int am1 = (wm * 32 + 16 + lidx) * 32;
    const int bn0 = (wn * 32 + lidx) * 32;
    const int bn1 = (wn * 32 + 16 + lidx) * 32;

    // prologue: fill buffer 0
    copy16_g2l(gA0, &As[0][lA0]);
    copy16_g2l(gA1, &As[0][lA1]);
    copy16_g2l(gW,  &Bs[0][lA0]);
    g2l_wave_fence();
    __syncthreads();

    int p = 0;
    for (int k0 = 0; k0 < K; k0 += 32) {
        // stream next K tile into the other buffer while we compute
        if (k0 + 32 < K) {
            copy16_g2l(gA0 + k0 + 32, &As[p ^ 1][lA0]);
            copy16_g2l(gA1 + k0 + 32, &As[p ^ 1][lA1]);
            copy16_g2l(gW  + k0 + 32, &Bs[p ^ 1][lA0]);
        }

        const _Float16* Asp = As[p];
        const _Float16* Bsp = Bs[p];
        v16h a0, a1, b0, b1;
        #pragma unroll
        for (int i = 0; i < 16; ++i) {
            int kk = ((i < 8) ? i : (i + 8)) + half_sel;
            a0[i] = Asp[am0 + kk];
            a1[i] = Asp[am1 + kk];
            b0[i] = Bsp[bn0 + kk];
            b1[i] = Bsp[bn1 + kk];
        }
        acc00 = __builtin_amdgcn_wmma_f32_16x16x32_f16(false, a0, false, b0, (short)0, acc00, false, false);
        acc01 = __builtin_amdgcn_wmma_f32_16x16x32_f16(false, a0, false, b1, (short)0, acc01, false, false);
        acc10 = __builtin_amdgcn_wmma_f32_16x16x32_f16(false, a1, false, b0, (short)0, acc10, false, false);
        acc11 = __builtin_amdgcn_wmma_f32_16x16x32_f16(false, a1, false, b1, (short)0, acc11, false, false);

        g2l_wave_fence();      // my async fills for buffer p^1 are done
        __syncthreads();       // everyone's fills visible; buffer p free
        p ^= 1;
    }

    // ---- epilogue ----
    const int rbase = m0 + wm * 32 + ((lane >> 4) << 3);
    const int cbase = n0 + wn * 32 + lidx;
    #pragma unroll
    for (int mi = 0; mi < 2; ++mi) {
        #pragma unroll
        for (int ni = 0; ni < 2; ++ni) {
            const v8f* accp = mi == 0 ? (ni == 0 ? &acc00 : &acc01)
                                      : (ni == 0 ? &acc10 : &acc11);
            #pragma unroll
            for (int r = 0; r < 8; ++r) {
                int row = rbase + mi * 16 + r;
                int col = cbase + ni * 16;
                float v = (*accp)[r];
                if (bias) v += bias[col];
                if (Cin)  v += Cin[(size_t)row * N + col];
                if (do_gelu) v = gelu_f(v);
                if (Cout)   Cout  [(size_t)row * N + col] = v;
                if (Cout16) Cout16[(size_t)row * N + col] = (_Float16)v;
            }
        }
    }
}

// ---------------------------------------------------------------------------
// Elementwise helpers
// ---------------------------------------------------------------------------
__global__ void f32_to_f16(const float* __restrict__ in, _Float16* __restrict__ out, int n) {
    int t = blockIdx.x * blockDim.x + threadIdx.x;
    if (t < n) out[t] = (_Float16)in[t];
}

__global__ void pad_w1_f16(const float* __restrict__ w1, _Float16* __restrict__ out) {
    int t = blockIdx.x * blockDim.x + threadIdx.x;   // 512*32
    if (t >= 512 * 32) return;
    int row = t >> 5, col = t & 31;
    out[t] = (col < 6) ? (_Float16)w1[row * 6 + col] : (_Float16)0.0f;
}

// Lifting layer 1: (32768,3) -> (32768,256), GELU, f16 out
__global__ void lift1_kernel(const float* __restrict__ x, const float* __restrict__ w1,
                             const float* __restrict__ b1, _Float16* __restrict__ h16) {
    int t = blockIdx.x * blockDim.x + threadIdx.x;   // 32768*256
    if (t >= 32768 * 256) return;
    int col = t & 255, row = t >> 8;
    float s = b1[col];
    #pragma unroll
    for (int c = 0; c < 3; ++c) s += x[row * 3 + c] * w1[col * 3 + c];
    h16[t] = (_Float16)gelu_f(s);
}

// ---------------------------------------------------------------------------
// Truncated DFT passes (only the 16x16x8 retained modes are ever computed).
// fft_norm='forward' => each forward pass carries 1/32; inverse carries 1.
// ---------------------------------------------------------------------------
__global__ void dft_fwd_z(const float* __restrict__ v, float* __restrict__ A1) {
    int t = blockIdx.x * blockDim.x + threadIdx.x;   // 32*32*8*64
    if (t >= 32 * 32 * 8 * 64) return;
    int ch = t & 63, kz = (t >> 6) & 7, y = (t >> 9) & 31, x = t >> 14;
    float sr = 0.f, si = 0.f;
    for (int z = 0; z < 32; ++z) {
        float val = v[((size_t)(x * 32 + y) * 32 + z) * 64 + ch];
        float s, c; sincosf(-TWO_PI * (float)(kz * z) / 32.0f, &s, &c);
        sr += val * c; si += val * s;
    }
    size_t o = ((size_t)(x * 32 + y) * 8 + kz) * 64 + ch;
    A1[o * 2 + 0] = sr * (1.0f / 32.0f);
    A1[o * 2 + 1] = si * (1.0f / 32.0f);
}

__global__ void dft_fwd_y(const float* __restrict__ A1, float* __restrict__ A2) {
    int t = blockIdx.x * blockDim.x + threadIdx.x;   // 32*16*8*64
    if (t >= 32 * 16 * 8 * 64) return;
    int ch = t & 63, kz = (t >> 6) & 7, kyj = (t >> 9) & 15, x = t >> 13;
    int ky = (kyj < 8) ? kyj : (kyj + 16);
    float sr = 0.f, si = 0.f;
    for (int y = 0; y < 32; ++y) {
        size_t idx = (((size_t)(x * 32 + y) * 8 + kz) * 64 + ch) * 2;
        float ar = A1[idx], ai = A1[idx + 1];
        float s, c; sincosf(TWO_PI * (float)(ky * y) / 32.0f, &s, &c);
        sr += ar * c + ai * s;      // * e^{-i th}
        si += ai * c - ar * s;
    }
    size_t o = (((size_t)(x * 16 + kyj) * 8 + kz) * 64 + ch) * 2;
    A2[o] = sr * (1.0f / 32.0f); A2[o + 1] = si * (1.0f / 32.0f);
}

__global__ void dft_fwd_x(const float* __restrict__ A2, float* __restrict__ VF) {
    int t = blockIdx.x * blockDim.x + threadIdx.x;   // 16*16*8*64
    if (t >= 16 * 16 * 8 * 64) return;
    int ch = t & 63, kz = (t >> 6) & 7, kyj = (t >> 9) & 15, kxj = t >> 13;
    int kx = (kxj < 8) ? kxj : (kxj + 16);
    float sr = 0.f, si = 0.f;
    for (int x = 0; x < 32; ++x) {
        size_t idx = (((size_t)(x * 16 + kyj) * 8 + kz) * 64 + ch) * 2;
        float ar = A2[idx], ai = A2[idx + 1];
        float s, c; sincosf(TWO_PI * (float)(kx * x) / 32.0f, &s, &c);
        sr += ar * c + ai * s;
        si += ai * c - ar * s;
    }
    size_t o = (((size_t)(kxj * 16 + kyj) * 8 + kz) * 64 + ch) * 2;
    VF[o] = sr * (1.0f / 32.0f); VF[o + 1] = si * (1.0f / 32.0f);
}

// Per-mode complex matvec: out[mode,o] = sum_i vf[mode,i] * w[c,i,o,mode]
__global__ void spectral_mul(const float* __restrict__ VF, const float* __restrict__ spec_w,
                             float* __restrict__ SO, int layer) {
    int t = blockIdx.x * blockDim.x + threadIdx.x;   // 4*512*64
    if (t >= 4 * 512 * 64) return;
    int o = t & 63, mode = (t >> 6) & 511, c = t >> 15;
    int mx = mode >> 6, my = (mode >> 3) & 7, mz = mode & 7;
    int kxj = mx + ((c & 2) ? 8 : 0);
    int kyj = my + ((c & 1) ? 8 : 0);
    size_t vf_base = ((size_t)(kxj * 16 + kyj) * 8 + mz) * 64;
    const float* wb = spec_w + ((((size_t)layer * 4 + c) * 64) * 64 + o) * 512 * 2
                             + (size_t)mode * 2;
    float sr = 0.f, si = 0.f;
    for (int i = 0; i < 64; ++i) {
        float ar = VF[(vf_base + i) * 2 + 0];
        float ai = VF[(vf_base + i) * 2 + 1];
        const float* wp = wb + (size_t)i * 64 * 512 * 2;
        float br = wp[0], bi = wp[1];
        sr += ar * br - ai * bi;
        si += ar * bi + ai * br;
    }
    SO[(vf_base + o) * 2 + 0] = sr;
    SO[(vf_base + o) * 2 + 1] = si;
}

__global__ void dft_inv_x(const float* __restrict__ SO, float* __restrict__ B1) {
    int t = blockIdx.x * blockDim.x + threadIdx.x;   // 32*16*8*64
    if (t >= 32 * 16 * 8 * 64) return;
    int o = t & 63, kz = (t >> 6) & 7, kyj = (t >> 9) & 15, x = t >> 13;
    float sr = 0.f, si = 0.f;
    for (int kxj = 0; kxj < 16; ++kxj) {
        int kx = (kxj < 8) ? kxj : (kxj + 16);
        size_t idx = (((size_t)(kxj * 16 + kyj) * 8 + kz) * 64 + o) * 2;
        float ar = SO[idx], ai = SO[idx + 1];
        float s, c; sincosf(TWO_PI * (float)(kx * x) / 32.0f, &s, &c);
        sr += ar * c - ai * s;      // * e^{+i th}
        si += ar * s + ai * c;
    }
    size_t oi = (((size_t)(x * 16 + kyj) * 8 + kz) * 64 + o) * 2;
    B1[oi] = sr; B1[oi + 1] = si;
}

__global__ void dft_inv_y(const float* __restrict__ B1, float* __restrict__ B2) {
    int t = blockIdx.x * blockDim.x + threadIdx.x;   // 32*32*8*64
    if (t >= 32 * 32 * 8 * 64) return;
    int o = t & 63, kz = (t >> 6) & 7, y = (t >> 9) & 31, x = t >> 14;
    float sr = 0.f, si = 0.f;
    for (int kyj = 0; kyj < 16; ++kyj) {
        int ky = (kyj < 8) ? kyj : (kyj + 16);
        size_t idx = (((size_t)(x * 16 + kyj) * 8 + kz) * 64 + o) * 2;
        float ar = B1[idx], ai = B1[idx + 1];
        float s, c; sincosf(TWO_PI * (float)(ky * y) / 32.0f, &s, &c);
        sr += ar * c - ai * s;
        si += ar * s + ai * c;
    }
    size_t oi = (((size_t)(x * 32 + y) * 8 + kz) * 64 + o) * 2;
    B2[oi] = sr; B2[oi + 1] = si;
}

// Hermitian irfft along z from 8 retained modes
__global__ void dft_inv_z(const float* __restrict__ B2, float* __restrict__ vnew) {
    int t = blockIdx.x * blockDim.x + threadIdx.x;   // 32*32*32*64
    if (t >= 32 * 32 * 32 * 64) return;
    int o = t & 63, z = (t >> 6) & 31, y = (t >> 11) & 31, x = t >> 16;
    size_t base = ((size_t)(x * 32 + y) * 8) * 64 + o;
    float acc = B2[base * 2];   // Re(C0)
    for (int k = 1; k < 8; ++k) {
        size_t idx = (base + (size_t)k * 64) * 2;
        float s, c; sincosf(TWO_PI * (float)(z * k) / 32.0f, &s, &c);
        acc += 2.0f * (B2[idx] * c - B2[idx + 1] * s);
    }
    vnew[((size_t)(x * 32 + y) * 32 + z) * 64 + o] = acc;
}

// ---------------------------------------------------------------------------
// GNO: build padded pair features + mask/index metadata for a query chunk
// ---------------------------------------------------------------------------
__global__ void build_pairs(const float* __restrict__ q, int q0,
                            _Float16* __restrict__ pair16,
                            int* __restrict__ mask, int* __restrict__ flat) {
    int t = blockIdx.x * blockDim.x + threadIdx.x;   // QC*27
    if (t >= PPC) return;
    int qi = q0 + t / 27, j = t % 27;
    const float h = 1.0f / 31.0f;
    float qx = q[qi * 3 + 0], qy = q[qi * 3 + 1], qz = q[qi * 3 + 2];
    int bx = (int)roundf(qx / h), by = (int)roundf(qy / h), bz = (int)roundf(qz / h);
    int ix = bx + (j / 9) - 1;
    int iy = by + ((j / 3) % 3) - 1;
    int iz = bz + (j % 3) - 1;
    bool inb = (ix >= 0) && (ix < 32) && (iy >= 0) && (iy < 32) && (iz >= 0) && (iz < 32);
    ix = min(max(ix, 0), 31); iy = min(max(iy, 0), 31); iz = min(max(iz, 0), 31);
    float yx = ix * h, yy = iy * h, yz = iz * h;
    float dx = qx - yx, dy = qy - yy, dz = qz - yz;
    float dist = sqrtf(dx * dx + dy * dy + dz * dz);
    mask[t] = (inb && dist <= 0.033f) ? 1 : 0;
    flat[t] = (ix * 32 + iy) * 32 + iz;
    _Float16* row = pair16 + (size_t)t * 32;
    row[0] = (_Float16)yx; row[1] = (_Float16)yy; row[2] = (_Float16)yz;
    row[3] = (_Float16)qx; row[4] = (_Float16)qy; row[5] = (_Float16)qz;
    #pragma unroll
    for (int i = 6; i < 32; ++i) row[i] = (_Float16)0.0f;
}

// masked mean over 27 neighbors: out[q,o] = sum(k*fv)/max(cnt,1)
__global__ void gno_reduce(const float* __restrict__ kbuf, const float* __restrict__ f_y,
                           const int* __restrict__ mask, const int* __restrict__ flat,
                           float* __restrict__ out, int q0) {
    int t = blockIdx.x * blockDim.x + threadIdx.x;   // QC*64
    if (t >= QC * 64) return;
    int o = t & 63, qi = t >> 6;
    float acc = 0.f; int cnt = 0;
    for (int j = 0; j < 27; ++j) {
        int r = qi * 27 + j;
        if (mask[r]) {
            ++cnt;
            acc += kbuf[(size_t)r * 64 + o] * f_y[(size_t)flat[r] * 64 + o];
        }
    }
    out[(size_t)(q0 + qi) * 64 + o] = acc / (float)max(cnt, 1);
}

// ---------------------------------------------------------------------------
extern "C" void kernel_launch(void* const* d_in, const int* in_sizes, int n_in,
                              void* d_out, int out_size, void* d_ws, size_t ws_size,
                              hipStream_t stream) {
    const float* x        = (const float*)d_in[0];
    const float* queries  = (const float*)d_in[1];
    const float* lift_w1  = (const float*)d_in[2];
    const float* lift_b1  = (const float*)d_in[3];
    const float* lift_w2  = (const float*)d_in[4];
    const float* lift_b2  = (const float*)d_in[5];
    const float* spec_w   = (const float*)d_in[6];
    const float* skip_w   = (const float*)d_in[7];
    const float* gno_w1   = (const float*)d_in[8];
    const float* gno_b1   = (const float*)d_in[9];
    const float* gno_w2   = (const float*)d_in[10];
    const float* gno_b2   = (const float*)d_in[11];
    const float* gno_w3   = (const float*)d_in[12];
    const float* gno_b3   = (const float*)d_in[13];
    float* out = (float*)d_out;

    unsigned char* base = (unsigned char*)d_ws;
    size_t off = 0;
    auto alloc = [&](size_t bytes) -> void* {
        void* p = base + off;
        off = (off + bytes + 255) & ~(size_t)255;
        return p;
    };
    const size_t NPTS = 32768;
    float*    v      = (float*)   alloc(NPTS * 64 * 4);
    float*    vnew   = (float*)   alloc(NPTS * 64 * 4);
    _Float16* v16    = (_Float16*)alloc(NPTS * 64 * 2);
    _Float16* h16    = (_Float16*)alloc(NPTS * 256 * 2);
    float*    A1     = (float*)   alloc((size_t)32 * 32 * 8 * 64 * 2 * 4);
    float*    A2     = (float*)   alloc((size_t)32 * 16 * 8 * 64 * 2 * 4);
    float*    VF     = (float*)   alloc((size_t)16 * 16 * 8 * 64 * 2 * 4);
    float*    SO     = (float*)   alloc((size_t)16 * 16 * 8 * 64 * 2 * 4);
    float*    B1     = (float*)   alloc((size_t)32 * 16 * 8 * 64 * 2 * 4);
    float*    B2     = (float*)   alloc((size_t)32 * 32 * 8 * 64 * 2 * 4);
    _Float16* wl2    = (_Float16*)alloc(64 * 256 * 2);
    _Float16* wsk    = (_Float16*)alloc(4 * 64 * 64 * 2);
    _Float16* wg1p   = (_Float16*)alloc(512 * 32 * 2);
    _Float16* wg2    = (_Float16*)alloc(256 * 512 * 2);
    _Float16* wg3    = (_Float16*)alloc(64 * 256 * 2);
    _Float16* pair16 = (_Float16*)alloc((size_t)PPC * 32 * 2);
    int*      maskb  = (int*)     alloc((size_t)PPC * 4);
    int*      flatb  = (int*)     alloc((size_t)PPC * 4);
    _Float16* h1b    = (_Float16*)alloc((size_t)PPC * 512 * 2);
    _Float16* h2b    = (_Float16*)alloc((size_t)PPC * 256 * 2);
    float*    kbuf   = (float*)   alloc((size_t)PPC * 64 * 4);
    (void)ws_size; (void)in_sizes; (void)n_in; (void)out_size;

    auto blocks = [](size_t n) { return (unsigned)((n + 255) / 256); };

    // --- one-time weight conversions to f16 ---
    f32_to_f16<<<blocks(64 * 256), 256, 0, stream>>>(lift_w2, wl2, 64 * 256);
    f32_to_f16<<<blocks(4 * 64 * 64), 256, 0, stream>>>(skip_w, wsk, 4 * 64 * 64);
    pad_w1_f16<<<blocks(512 * 32), 256, 0, stream>>>(gno_w1, wg1p);
    f32_to_f16<<<blocks(256 * 512), 256, 0, stream>>>(gno_w2, wg2, 256 * 512);
    f32_to_f16<<<blocks(64 * 256), 256, 0, stream>>>(gno_w3, wg3, 64 * 256);

    // --- lifting: 3 -> 256 (gelu) -> 64 ---
    lift1_kernel<<<blocks((size_t)NPTS * 256), 256, 0, stream>>>(x, lift_w1, lift_b1, h16);
    gemm_wmma<<<dim3(64 / 64, NPTS / 128), 256, 0, stream>>>(
        h16, wl2, lift_b2, nullptr, v, nullptr, (int)NPTS, 64, 256, 0);

    // --- 4 FNO blocks ---
    for (int l = 0; l < 4; ++l) {
        f32_to_f16<<<blocks(NPTS * 64), 256, 0, stream>>>(v, v16, (int)(NPTS * 64));
        dft_fwd_z<<<blocks((size_t)32 * 32 * 8 * 64), 256, 0, stream>>>(v, A1);
        dft_fwd_y<<<blocks((size_t)32 * 16 * 8 * 64), 256, 0, stream>>>(A1, A2);
        dft_fwd_x<<<blocks((size_t)16 * 16 * 8 * 64), 256, 0, stream>>>(A2, VF);
        spectral_mul<<<blocks((size_t)4 * 512 * 64), 256, 0, stream>>>(VF, spec_w, SO, l);
        dft_inv_x<<<blocks((size_t)32 * 16 * 8 * 64), 256, 0, stream>>>(SO, B1);
        dft_inv_y<<<blocks((size_t)32 * 32 * 8 * 64), 256, 0, stream>>>(B1, B2);
        dft_inv_z<<<blocks((size_t)32 * 32 * 32 * 64), 256, 0, stream>>>(B2, vnew);
        // v = gelu?(vnew + v @ skip^T)
        gemm_wmma<<<dim3(1, NPTS / 128), 256, 0, stream>>>(
            v16, wsk + (size_t)l * 64 * 64, nullptr, vnew, v, nullptr,
            (int)NPTS, 64, 64, (l < 3) ? 1 : 0);
    }

    // --- GNO output: chunked 3-layer MLP GEMMs + masked mean ---
    for (int ch = 0; ch < NQ / QC; ++ch) {
        int q0 = ch * QC;
        build_pairs<<<blocks(PPC), 256, 0, stream>>>(queries, q0, pair16, maskb, flatb);
        gemm_wmma<<<dim3(512 / 64, PPC / 128), 256, 0, stream>>>(
            pair16, wg1p, gno_b1, nullptr, nullptr, h1b, PPC, 512, 32, 1);
        gemm_wmma<<<dim3(256 / 64, PPC / 128), 256, 0, stream>>>(
            h1b, wg2, gno_b2, nullptr, nullptr, h2b, PPC, 256, 512, 1);
        gemm_wmma<<<dim3(64 / 64, PPC / 128), 256, 0, stream>>>(
            h2b, wg3, gno_b3, nullptr, kbuf, nullptr, PPC, 64, 256, 0);
        gno_reduce<<<blocks((size_t)QC * 64), 256, 0, stream>>>(
            kbuf, v, maskb, flatb, out, q0);
    }
}
